// TransformerEncoderLayer_43576738185916
// MI455X (gfx1250) — compile-verified, matrix operand order checked by reading, not run on hardware
//
#include <hip/hip_runtime.h>
#include <hip/hip_bf16.h>
#include <math.h>

typedef _Float16 half_t;
typedef __attribute__((ext_vector_type(16))) _Float16 v16h;
typedef __attribute__((ext_vector_type(8)))  _Float16 v8h;
typedef __attribute__((ext_vector_type(8)))  float    v8f;

#define T_DIM 2048
#define N_DIM 4
#define C_DIM 1024
#define H_DIM 16
#define D_DIM 64
#define W_DIM 512
#define F_DIM 4096
#define NT    (T_DIM * N_DIM)

#define BM 64
#define BN 128
#define BK 32
#define LSTR 40   // padded LDS stride in halves (80B: 16B aligned, avoids bank conflicts)

enum { MODE_F16_BIAS = 0, MODE_F16_GELU = 1, MODE_F32_RES_T = 2, MODE_F32_OUT_T = 3 };

__device__ __forceinline__ v16h make_frag(v8h lo, v8h hi) {
  v16h a;
#pragma unroll
  for (int i = 0; i < 8; i++) { a[i] = lo[i]; a[i + 8] = hi[i]; }
  return a;
}

__device__ __forceinline__ v8f wmma_f16(v16h a, v16h b, v8f c) {
  return __builtin_amdgcn_wmma_f32_16x16x32_f16(false, a, false, b, (short)0, c, false, false);
}

// ---------------- elementwise: f32 -> f16 cast ----------------
__global__ void cast_f32_to_f16(const float* __restrict__ src, half_t* __restrict__ dst, int n) {
  int i = blockIdx.x * blockDim.x + threadIdx.x;
  if (i < n) dst[i] = (half_t)src[i];
}

// ---------------- LayerNorm (one block per row), writes f16 ----------------
template <bool TRANSPOSED>
__global__ __launch_bounds__(256) void layernorm_f16(const float* __restrict__ src,
                                                     const float* __restrict__ g,
                                                     const float* __restrict__ b,
                                                     half_t* __restrict__ dst) {
  int r = blockIdx.x;  // r = n*T + t
  const float* row;
  if (TRANSPOSED) {
    int n = r >> 11;             // T = 2048
    int t = r & (T_DIM - 1);
    row = src + ((size_t)t * N_DIM + n) * C_DIM;
  } else {
    row = src + (size_t)r * C_DIM;
  }
  float v[4];
  float s = 0.f, s2 = 0.f;
#pragma unroll
  for (int i = 0; i < 4; i++) {
    v[i] = row[threadIdx.x + i * 256];
    s += v[i];
    s2 += v[i] * v[i];
  }
#pragma unroll
  for (int m = 16; m >= 1; m >>= 1) { s += __shfl_xor(s, m); s2 += __shfl_xor(s2, m); }
  __shared__ float sh[2][8];
  int wave = threadIdx.x >> 5, lane = threadIdx.x & 31;
  if (lane == 0) { sh[0][wave] = s; sh[1][wave] = s2; }
  __syncthreads();
  if (wave == 0) {
    s = sh[0][lane & 7];
    s2 = sh[1][lane & 7];
#pragma unroll
    for (int m = 4; m >= 1; m >>= 1) { s += __shfl_xor(s, m); s2 += __shfl_xor(s2, m); }
    if (lane == 0) { sh[0][0] = s; sh[1][0] = s2; }
  }
  __syncthreads();
  float mu = sh[0][0] * (1.0f / C_DIM);
  float var = sh[1][0] * (1.0f / C_DIM) - mu * mu;
  float rstd = rsqrtf(var + 1e-5f);
#pragma unroll
  for (int i = 0; i < 4; i++) {
    int c = threadIdx.x + i * 256;
    dst[(size_t)r * C_DIM + c] = (half_t)((v[i] - mu) * rstd * g[c] + b[c]);
  }
}

// ---------------- RoPE, in place on f16 (pos = t + W) ----------------
__global__ void rope_inplace(half_t* __restrict__ buf, int stride) {
  int idx = blockIdx.x * blockDim.x + threadIdx.x;
  if (idx >= NT * H_DIM * (D_DIM / 2)) return;
  int i = idx & 31;
  int h = (idx >> 5) & 15;
  int row = idx >> 9;
  int t = row & (T_DIM - 1);
  float pos = (float)(t + W_DIM);
  // inv_freq = ROPE_BASE^(-2i/D) = exp(-(2i/D) * ln(10000))
  float freq = __expf(-(float)(2 * i) * (1.0f / (float)D_DIM) * 9.210340371976184f);
  float ang = pos * freq;
  float sn, cs;
  __sincosf(ang, &sn, &cs);
  half_t* p = buf + (size_t)row * stride + h * D_DIM;
  float a = (float)p[i];
  float bq = (float)p[i + 32];
  p[i] = (half_t)(a * cs - bq * sn);
  p[i + 32] = (half_t)(bq * cs + a * sn);
}

// ---------------- WMMA GEMM: C[M x Ncols] = A(f16, MxK) @ B(f16, KxNcols) + epilogue ----
// Block tile 64x128, 8 waves; each wave owns a 32x32 macro-tile:
// 2 A-frags x 2 B-frags -> 4 WMMAs per K-step (2x the WMMA:LDS-load density of 64x64).
template <int MODE>
__global__ __launch_bounds__(256) void gemm_wmma(const half_t* __restrict__ A,
                                                 const half_t* __restrict__ B,
                                                 const float* __restrict__ bias,
                                                 const float* __restrict__ res,
                                                 half_t* __restrict__ out16,
                                                 float* __restrict__ out32,
                                                 int K, int Ncols) {
  __shared__ __align__(16) half_t sA[BM * LSTR];   // 64 x 32 (padded)
  __shared__ __align__(16) half_t sB[BN * LSTR];   // transposed: [n][k], 128 x 32 (padded)

  const int tid = threadIdx.x;
  const int lane = tid & 31;
  const int wave = tid >> 5;
  const int wm = wave >> 2;  // 0..1 : 32-row strip
  const int wn = wave & 3;   // 0..3 : 32-col strip
  const int bm = blockIdx.x * BM;
  const int bn = blockIdx.y * BN;

  const int kb  = (lane & 16) ? 8 : 0;
  const int l15 = lane & 15;
  const int hi8 = (lane & 16) ? 8 : 0;

  v8f acc[2][2];
#pragma unroll
  for (int a2 = 0; a2 < 2; a2++)
#pragma unroll
    for (int b2 = 0; b2 < 2; b2++)
#pragma unroll
      for (int j = 0; j < 8; j++) acc[a2][b2][j] = 0.f;

  const int arow = tid >> 2;        // 0..63
  const int akc  = (tid & 3) * 8;   // 0,8,16,24
  const int brow = tid >> 4;        // 0..15 (two passes cover k=0..31)
  const int bc8  = (tid & 15) * 8;  // 0..120

  for (int kt = 0; kt < K; kt += BK) {
    // stage A (64x32): one b128 per thread
    v8h av = *(const v8h*)(A + (size_t)(bm + arow) * K + kt + akc);
    *(v8h*)(sA + arow * LSTR + akc) = av;
    // stage B (32x128) transposed into sB[n][k]: two b128 per thread
#pragma unroll
    for (int p = 0; p < 2; p++) {
      int k = brow + p * 16;
      v8h bv = *(const v8h*)(B + (size_t)(kt + k) * Ncols + bn + bc8);
#pragma unroll
      for (int i = 0; i < 8; i++) sB[(bc8 + i) * LSTR + k] = bv[i];
    }
    // prefetch next K-tile into cache while WMMAs run (global_prefetch_b8)
    if (kt + BK < K) {
      __builtin_prefetch(A + (size_t)(bm + arow) * K + kt + BK + akc, 0, 1);
      __builtin_prefetch(B + (size_t)(kt + BK + brow) * Ncols + bn + bc8, 0, 1);
      __builtin_prefetch(B + (size_t)(kt + BK + brow + 16) * Ncols + bn + bc8, 0, 1);
    }
    __syncthreads();

    v16h af[2], bf[2];
#pragma unroll
    for (int a2 = 0; a2 < 2; a2++) {
      const half_t* pa = sA + (wm * 32 + a2 * 16 + l15) * LSTR;
      af[a2] = make_frag(*(const v8h*)(pa + kb), *(const v8h*)(pa + kb + 16));
    }
#pragma unroll
    for (int b2 = 0; b2 < 2; b2++) {
      const half_t* pb = sB + (wn * 32 + b2 * 16 + l15) * LSTR;
      bf[b2] = make_frag(*(const v8h*)(pb + kb), *(const v8h*)(pb + kb + 16));
    }
#pragma unroll
    for (int a2 = 0; a2 < 2; a2++)
#pragma unroll
      for (int b2 = 0; b2 < 2; b2++)
        acc[a2][b2] = wmma_f16(af[a2], bf[b2], acc[a2][b2]);
    __syncthreads();
  }

#pragma unroll
  for (int a2 = 0; a2 < 2; a2++) {
    const int r0 = bm + wm * 32 + a2 * 16 + hi8;
#pragma unroll
    for (int b2 = 0; b2 < 2; b2++) {
      const int c0 = bn + wn * 32 + b2 * 16 + l15;
      const float bc = bias[c0];
#pragma unroll
      for (int j = 0; j < 8; j++) {
        int r = r0 + j;
        float v0 = acc[a2][b2][j] + bc;
        if constexpr (MODE == MODE_F16_BIAS) {
          out16[(size_t)r * Ncols + c0] = (half_t)v0;
        } else if constexpr (MODE == MODE_F16_GELU) {
          v0 = 0.5f * v0 * (1.0f + erff(v0 * 0.70710678118f));
          out16[(size_t)r * Ncols + c0] = (half_t)v0;
        } else if constexpr (MODE == MODE_F32_RES_T) {
          // h = x(transposed read) + proj
          int n = r >> 11, t = r & (T_DIM - 1);
          size_t tr = ((size_t)t * N_DIM + n) * C_DIM;
          out32[(size_t)r * C_DIM + c0] = v0 + res[tr + c0];
        } else {  // MODE_F32_OUT_T : d_out[(t,n,c)] = h + ffn
          int n = r >> 11, t = r & (T_DIM - 1);
          size_t tr = ((size_t)t * N_DIM + n) * C_DIM;
          out32[tr + c0] = v0 + res[(size_t)r * C_DIM + c0];
        }
      }
    }
  }
}

// ---------------- Banded flash attention: one wave per (n, h, 16-query tile) ----
__global__ __launch_bounds__(32) void attn_kernel(const half_t* __restrict__ Q,
                                                  const half_t* __restrict__ KV,
                                                  half_t* __restrict__ O) {
  __shared__ __align__(16) half_t sP[16 * LSTR];
  __shared__ __align__(16) half_t sV[64 * LSTR];  // V transposed: [d][key]

  const int lane = threadIdx.x;
  const int q0 = blockIdx.x * 16;
  const int h  = blockIdx.y;
  const int n  = blockIdx.z;

  const int kb  = (lane & 16) ? 8 : 0;
  const int l15 = lane & 15;
  const int hi8 = (lane & 16) ? 8 : 0;

  const half_t* Qb = Q + ((size_t)(n * T_DIM + q0)) * C_DIM + h * D_DIM;
  const half_t* Kb = KV + ((size_t)n * T_DIM) * (2 * C_DIM) + h * D_DIM;
  const half_t* Vb = Kb + C_DIM;

  // Q as A-fragments for both 32-wide D steps (per-lane runs are contiguous in D)
  const half_t* qr = Qb + (size_t)l15 * C_DIM;
  v16h aQ0 = make_frag(*(const v8h*)(qr + kb), *(const v8h*)(qr + kb + 16));
  v16h aQ1 = make_frag(*(const v8h*)(qr + 32 + kb), *(const v8h*)(qr + 32 + kb + 16));

  float m[8], l[8];
  v8f oo[4];
#pragma unroll
  for (int j = 0; j < 8; j++) { m[j] = -3e38f; l[j] = 0.f; }
#pragma unroll
  for (int dt = 0; dt < 4; dt++)
#pragma unroll
    for (int j = 0; j < 8; j++) oo[dt][j] = 0.f;

  int kstart = q0 - W_DIM;
  if (kstart < 0) kstart = 0;
  kstart &= ~31;

  for (int kc = kstart; kc <= q0 + 15; kc += 32) {
    // K B-fragments (column = key, contiguous runs in D per lane)
    const half_t* kr0 = Kb + (size_t)(kc + l15) * (2 * C_DIM);
    const half_t* kr1 = kr0 + (size_t)16 * (2 * C_DIM);
    v16h bK00 = make_frag(*(const v8h*)(kr0 + kb), *(const v8h*)(kr0 + kb + 16));
    v16h bK01 = make_frag(*(const v8h*)(kr0 + 32 + kb), *(const v8h*)(kr0 + 32 + kb + 16));
    v16h bK10 = make_frag(*(const v8h*)(kr1 + kb), *(const v8h*)(kr1 + kb + 16));
    v16h bK11 = make_frag(*(const v8h*)(kr1 + 32 + kb), *(const v8h*)(kr1 + 32 + kb + 16));

    v8f s0 = {0, 0, 0, 0, 0, 0, 0, 0};
    v8f s1 = {0, 0, 0, 0, 0, 0, 0, 0};
    s0 = wmma_f16(aQ0, bK00, s0);
    s0 = wmma_f16(aQ1, bK01, s0);
    s1 = wmma_f16(aQ0, bK10, s1);
    s1 = wmma_f16(aQ1, bK11, s1);

    // stage V chunk transposed into LDS: sV[d][key]
    {
      const half_t* vr = Vb + (size_t)(kc + lane) * (2 * C_DIM);
#pragma unroll
      for (int g = 0; g < 8; g++) {
        v8h vv = *(const v8h*)(vr + g * 8);
#pragma unroll
        for (int i2 = 0; i2 < 8; i2++) sV[(g * 8 + i2) * LSTR + lane] = vv[i2];
      }
    }

    // online softmax (band mask as data-level select; EXEC stays all-1s)
    float pv0[8], pv1[8];
#pragma unroll
    for (int j = 0; j < 8; j++) {
      int q = q0 + j + hi8;
      int k0i = kc + l15, k1i = kc + 16 + l15;
      bool val0 = (k0i <= q) && (q - k0i <= W_DIM);
      bool val1 = (k1i <= q) && (q - k1i <= W_DIM);
      float x0 = val0 ? s0[j] * 0.125f : -3e38f;
      float x1 = val1 ? s1[j] * 0.125f : -3e38f;
      float cm = fmaxf(x0, x1);
#pragma unroll
      for (int msk = 1; msk < 16; msk <<= 1) cm = fmaxf(cm, __shfl_xor(cm, msk));
      float nm = fmaxf(m[j], cm);
      float corr = __expf(m[j] - nm);  // (-3e38)-(-3e38)=0 -> 1; finite path ok
      m[j] = nm;
      float p0 = val0 ? __expf(x0 - nm) : 0.f;
      float p1 = val1 ? __expf(x1 - nm) : 0.f;
      float rs = p0 + p1;
#pragma unroll
      for (int msk = 1; msk < 16; msk <<= 1) rs += __shfl_xor(rs, msk);
      l[j] = l[j] * corr + rs;
#pragma unroll
      for (int dt = 0; dt < 4; dt++) oo[dt][j] *= corr;
      pv0[j] = p0;
      pv1[j] = p1;
    }

    // P -> LDS (re-layout from C/D layout to A-fragment layout)
#pragma unroll
    for (int j = 0; j < 8; j++) {
      sP[(j + hi8) * LSTR + l15] = (half_t)pv0[j];
      sP[(j + hi8) * LSTR + 16 + l15] = (half_t)pv1[j];
    }
    __syncthreads();  // single-wave WG: lowers to NOP, LDS waits still enforced

    const half_t* pp = sP + l15 * LSTR;
    v16h aP = make_frag(*(const v8h*)(pp + kb), *(const v8h*)(pp + kb + 16));
#pragma unroll
    for (int dt = 0; dt < 4; dt++) {
      const half_t* pv = sV + (dt * 16 + l15) * LSTR;
      v16h bV = make_frag(*(const v8h*)(pv + kb), *(const v8h*)(pv + kb + 16));
      oo[dt] = wmma_f16(aP, bV, oo[dt]);
    }
    __syncthreads();
  }

  float inv[8];
#pragma unroll
  for (int j = 0; j < 8; j++) inv[j] = (l[j] > 0.f) ? (1.0f / l[j]) : 0.f;

  half_t* ob = O + ((size_t)(n * T_DIM + q0)) * C_DIM + h * D_DIM;
#pragma unroll
  for (int j = 0; j < 8; j++) {
    size_t rb = (size_t)(j + hi8) * C_DIM;
#pragma unroll
    for (int dt = 0; dt < 4; dt++)
      ob[rb + dt * 16 + l15] = (half_t)(oo[dt][j] * inv[j]);
  }
}

// ---------------- host orchestration ----------------
extern "C" void kernel_launch(void* const* d_in, const int* in_sizes, int n_in,
                              void* d_out, int out_size, void* d_ws, size_t ws_size,
                              hipStream_t stream) {
  (void)in_sizes; (void)n_in; (void)out_size; (void)ws_size;
  const float* x     = (const float*)d_in[0];
  const float* q_w   = (const float*)d_in[1];
  const float* q_b   = (const float*)d_in[2];
  const float* kv_w  = (const float*)d_in[3];
  const float* kv_b  = (const float*)d_in[4];
  const float* out_w = (const float*)d_in[5];
  const float* out_b = (const float*)d_in[6];
  const float* ff_w1 = (const float*)d_in[7];
  const float* ff_b1 = (const float*)d_in[8];
  const float* ff_w2 = (const float*)d_in[9];
  const float* ff_b2 = (const float*)d_in[10];
  const float* ln1_g = (const float*)d_in[11];
  const float* ln1_b = (const float*)d_in[12];
  const float* ln2_g = (const float*)d_in[13];
  const float* ln2_b = (const float*)d_in[14];

  char* ws = (char*)d_ws;
  size_t off = 0;
  auto alloc = [&](size_t bytes) -> char* {
    char* p = ws + off;
    off += (bytes + 255) & ~(size_t)255;
    return p;
  };
  half_t* wq   = (half_t*)alloc((size_t)C_DIM * C_DIM * 2);
  half_t* wkv  = (half_t*)alloc((size_t)C_DIM * 2 * C_DIM * 2);
  half_t* wo   = (half_t*)alloc((size_t)C_DIM * C_DIM * 2);
  half_t* wf1  = (half_t*)alloc((size_t)C_DIM * F_DIM * 2);
  half_t* wf2  = (half_t*)alloc((size_t)F_DIM * C_DIM * 2);
  half_t* A16  = (half_t*)alloc((size_t)NT * C_DIM * 2);
  half_t* Q16  = (half_t*)alloc((size_t)NT * C_DIM * 2);
  half_t* KV16 = (half_t*)alloc((size_t)NT * 2 * C_DIM * 2);
  half_t* O16  = (half_t*)alloc((size_t)NT * C_DIM * 2);
  float*  H32  = (float*)alloc((size_t)NT * C_DIM * 4);
  half_t* G16  = (half_t*)alloc((size_t)NT * F_DIM * 2);

  auto cast = [&](const float* s, half_t* d, int n) {
    cast_f32_to_f16<<<(n + 255) / 256, 256, 0, stream>>>(s, d, n);
  };
  cast(q_w, wq, C_DIM * C_DIM);
  cast(kv_w, wkv, C_DIM * 2 * C_DIM);
  cast(out_w, wo, C_DIM * C_DIM);
  cast(ff_w1, wf1, C_DIM * F_DIM);
  cast(ff_w2, wf2, F_DIM * C_DIM);

  layernorm_f16<true><<<NT, 256, 0, stream>>>(x, ln1_g, ln1_b, A16);

  dim3 blk(256);
  gemm_wmma<MODE_F16_BIAS><<<dim3(NT / BM, C_DIM / BN), blk, 0, stream>>>(
      A16, wq, q_b, nullptr, Q16, nullptr, C_DIM, C_DIM);
  gemm_wmma<MODE_F16_BIAS><<<dim3(NT / BM, 2 * C_DIM / BN), blk, 0, stream>>>(
      A16, wkv, kv_b, nullptr, KV16, nullptr, C_DIM, 2 * C_DIM);

  int nrope = NT * H_DIM * (D_DIM / 2);
  rope_inplace<<<(nrope + 255) / 256, 256, 0, stream>>>(Q16, C_DIM);
  rope_inplace<<<(nrope + 255) / 256, 256, 0, stream>>>(KV16, 2 * C_DIM);

  attn_kernel<<<dim3(T_DIM / 16, H_DIM, N_DIM), 32, 0, stream>>>(Q16, KV16, O16);

  gemm_wmma<MODE_F32_RES_T><<<dim3(NT / BM, C_DIM / BN), blk, 0, stream>>>(
      O16, wo, out_b, x, nullptr, H32, C_DIM, C_DIM);

  layernorm_f16<false><<<NT, 256, 0, stream>>>(H32, ln2_g, ln2_b, A16);

  gemm_wmma<MODE_F16_GELU><<<dim3(NT / BM, F_DIM / BN), blk, 0, stream>>>(
      A16, wf1, ff_b1, nullptr, G16, nullptr, C_DIM, F_DIM);

  gemm_wmma<MODE_F32_OUT_T><<<dim3(NT / BM, C_DIM / BN), blk, 0, stream>>>(
      G16, wf2, ff_b2, H32, nullptr, (float*)d_out, F_DIM, C_DIM);
}
